// BigGAT_5918464934680
// MI455X (gfx1250) — compile-verified, matrix operand order checked by reading, not run on hardware
//
#include <hip/hip_runtime.h>
#include <stdint.h>

#define NNODES 50000
#define NEDGES 800000
#define HEADS  4
#define FDIM   128          // IN == H*HID == 128 feature width at GEMM inputs
#define ROWTILES 3125       // 50000 / 16

typedef __bf16 bf16_t;
typedef bf16_t v16bf __attribute__((ext_vector_type(16)));
typedef float  v8f   __attribute__((ext_vector_type(8)));

union PackU { unsigned u[8]; v16bf v; };

// ---------- helpers ----------
__device__ __forceinline__ unsigned short f2bf(float f) {
    unsigned u = __float_as_uint(f);
    unsigned r = u + 0x7FFFu + ((u >> 16) & 1u);   // round-to-nearest-even
    return (unsigned short)(r >> 16);
}
__device__ __forceinline__ unsigned f2mono(float f) {  // order-preserving float->uint
    unsigned u = __float_as_uint(f);
    return (u & 0x80000000u) ? ~u : (u | 0x80000000u);
}
__device__ __forceinline__ float mono2f(unsigned u) {
    unsigned b = (u & 0x80000000u) ? (u ^ 0x80000000u) : ~u;
    return __uint_as_float(b);
}
__device__ __forceinline__ float lrelu(float x) { return x > 0.f ? x : 0.2f * x; }
__device__ __forceinline__ void edge_sd(const int* __restrict__ ei, int i, int& s, int& d) {
    if (i < NEDGES) { s = ei[i]; d = ei[NEDGES + i]; }
    else            { s = i - NEDGES; d = s; }          // self loops
}

// ---------- utility kernels ----------
__global__ void k_zero_u32(unsigned* __restrict__ p, int n) {
    int i = blockIdx.x * 256 + threadIdx.x;
    if (i < n) p[i] = 0u;
}
__global__ void k_f32_to_bf16(const float* __restrict__ in, unsigned short* __restrict__ out, int n) {
    int i = blockIdx.x * 256 + threadIdx.x;
    if (i < n) out[i] = f2bf(in[i]);
}

// Pack B = [Wmain | Wskip^T] (K=128 rows, ctot cols) into WMMA per-lane dword pairs.
// Layout: Bp[((tile*4 + kc)*32 + lane)*8 + v] = bf16 pair (K, K+1) for column n = tile*16 + lane%16
__global__ void k_pack_b(const float* __restrict__ Wmain, const float* __restrict__ Wskip,
                         unsigned* __restrict__ Bp, int cmain, int total_threads) {
    int tid = blockIdx.x * 256 + threadIdx.x;
    if (tid >= total_threads) return;
    int v    = tid & 7;
    int lane = (tid >> 3) & 31;
    int kc   = (tid >> 8) & 3;
    int t    = tid >> 10;
    int n = t * 16 + (lane & 15);
    int K = kc * 32 + ((v >> 2) << 4) + ((lane >> 4) << 3) + ((v & 3) << 1);
    float lo, hi;
    if (n < cmain) { lo = Wmain[(size_t)K * cmain + n];       hi = Wmain[(size_t)(K + 1) * cmain + n]; }
    else { int r = n - cmain; lo = Wskip[(size_t)r * FDIM + K]; hi = Wskip[(size_t)r * FDIM + K + 1]; }
    Bp[tid] = (unsigned)f2bf(lo) | ((unsigned)f2bf(hi) << 16);
}

// ---------- WMMA GEMM: [50000 x 128](bf16) x [128 x (c1+c2)](bf16) -> out1 [.,c1] f32, out2 [.,c2] f32
// One wave computes TWO adjacent 16x16 column tiles (A fragment loaded once, 8 WMMAs/wave).
// Requires (c1+c2)/16 even and c1 a multiple of 16 (true for both layers: 16 and 10 tiles, c1=128).
__global__ void k_gemm_wmma(const unsigned short* __restrict__ A, const unsigned* __restrict__ Bp,
                            float* __restrict__ out1, float* __restrict__ out2, int c1, int c2) {
    int lane = threadIdx.x & 31;
    int wid  = threadIdx.x >> 5;
    int ntiles = (c1 + c2) >> 4;
    int ct0 = wid * 2;                              // wave-uniform pair of column tiles
    if (ct0 >= ntiles) return;                      // whole-wave exit, EXEC stays all-1s
    int ct1 = ct0 + 1;
    int m0  = blockIdx.x * 16;
    int row = m0 + (lane & 15);
    const unsigned short* Ar = A + (size_t)row * FDIM;

    v8f acc0 = {}, acc1 = {};
#pragma unroll
    for (int kc = 0; kc < 4; ++kc) {
        PackU a, b0, b1;
#pragma unroll
        for (int v = 0; v < 8; ++v) {
            int K = kc * 32 + ((v >> 2) << 4) + ((lane >> 4) << 3) + ((v & 3) << 1);
            a.u[v] = *(const unsigned*)(Ar + K);
        }
        const unsigned* p0 = Bp + (((size_t)ct0 * 4 + kc) * 32 + lane) * 8;
        const unsigned* p1 = Bp + (((size_t)ct1 * 4 + kc) * 32 + lane) * 8;
#pragma unroll
        for (int v = 0; v < 8; ++v) { b0.u[v] = p0[v]; b1.u[v] = p1[v]; }
        acc0 = __builtin_amdgcn_wmma_f32_16x16x32_bf16(
            false, a.v, false, b0.v, (short)0, acc0, false, false);
        acc1 = __builtin_amdgcn_wmma_f32_16x16x32_bf16(
            false, a.v, false, b1.v, (short)0, acc1, false, false);
    }

    int nloc  = lane & 15;
    int mbase = (lane >> 4) << 3;                   // VGPR r -> M = r (lanes 0-15), 8+r (lanes 16-31)
#pragma unroll
    for (int t = 0; t < 2; ++t) {
        int ct = t ? ct1 : ct0;
        const v8f& acc = t ? acc1 : acc0;
        // tile is entirely in out1 or out2 (c1 % 16 == 0): wave-uniform select
        bool in1 = (ct * 16) < c1;
        float* dst = in1 ? out1 : out2;
        int ld  = in1 ? c1 : c2;
        int col = ct * 16 + nloc - (in1 ? 0 : c1);
#pragma unroll
        for (int r = 0; r < 8; ++r)
            dst[(size_t)(m0 + mbase + r) * ld + col] = acc[r];
    }
}

// ---------- per-node attention logits: es/ed [N,4]
__global__ void k_attn(const float* __restrict__ h, const float* __restrict__ asrc,
                       const float* __restrict__ adst, float* __restrict__ es, float* __restrict__ ed) {
    int t = blockIdx.x * 256 + threadIdx.x;
    if (t >= NNODES * HEADS) return;
    int n = t >> 2, hd = t & 3;
    const float* hr = h + (size_t)n * FDIM + hd * 32;
    const float* as = asrc + hd * 32;
    const float* ad = adst + hd * 32;
    float s = 0.f, d = 0.f;
#pragma unroll
    for (int c = 0; c < 32; ++c) { float v = hr[c]; s += v * as[c]; d += v * ad[c]; }
    es[t] = s; ed[t] = d;
}

// ---------- edge pass 1: scatter-max of logits into m[dst,head] (monotone-uint atomic max)
__global__ void k_edge_max(const int* __restrict__ ei, const float* __restrict__ es,
                           const float* __restrict__ ed, unsigned* __restrict__ m) {
    int i = blockIdx.x * 256 + threadIdx.x;
    if (i >= NEDGES + NNODES) return;
    int s, d; edge_sd(ei, i, s, d);
    float4 esv = *(const float4*)(es + (size_t)s * 4);
    float4 edv = *(const float4*)(ed + (size_t)d * 4);
    atomicMax(&m[d * 4 + 0], f2mono(lrelu(esv.x + edv.x)));
    atomicMax(&m[d * 4 + 1], f2mono(lrelu(esv.y + edv.y)));
    atomicMax(&m[d * 4 + 2], f2mono(lrelu(esv.z + edv.z)));
    atomicMax(&m[d * 4 + 3], f2mono(lrelu(esv.w + edv.w)));
}

// ---------- edge pass 2: softmax denominator
__global__ void k_edge_denom(const int* __restrict__ ei, const float* __restrict__ es,
                             const float* __restrict__ ed, const unsigned* __restrict__ m,
                             float* __restrict__ denom, const float* __restrict__ h) {
    int i = blockIdx.x * 256 + threadIdx.x;
    if (i >= NEDGES + NNODES) return;
    int s, d; edge_sd(ei, i, s, d);
    __builtin_prefetch(h + (size_t)s * FDIM, 0, 1);  // warm src feature row for scatter pass
    float4 esv = *(const float4*)(es + (size_t)s * 4);
    float4 edv = *(const float4*)(ed + (size_t)d * 4);
    const uint4 mv = *(const uint4*)(m + (size_t)d * 4);
    atomicAdd(&denom[d * 4 + 0], __expf(lrelu(esv.x + edv.x) - mono2f(mv.x)));
    atomicAdd(&denom[d * 4 + 1], __expf(lrelu(esv.y + edv.y) - mono2f(mv.y)));
    atomicAdd(&denom[d * 4 + 2], __expf(lrelu(esv.z + edv.z) - mono2f(mv.z)));
    atomicAdd(&denom[d * 4 + 3], __expf(lrelu(esv.w + edv.w) - mono2f(mv.w)));
}

// ---------- edge pass 3: wave per edge, gather h[src] (float4/lane), scatter alpha*h into aggr[dst]
__global__ void k_edge_scatter(const int* __restrict__ ei, const float* __restrict__ es,
                               const float* __restrict__ ed, const unsigned* __restrict__ m,
                               const float* __restrict__ denom, const float* __restrict__ h,
                               float* __restrict__ aggr) {
    int lane = threadIdx.x & 31;
    int e = blockIdx.x * 8 + (threadIdx.x >> 5);
    if (e >= NEDGES + NNODES) return;
    int s, d; edge_sd(ei, e, s, d);
    int hd = lane >> 3;                               // 8 lanes (32 ch) per head
    float lg = lrelu(es[s * 4 + hd] + ed[d * 4 + hd]);
    float w  = __expf(lg - mono2f(m[d * 4 + hd]));
    float a  = w / (denom[d * 4 + hd] + 1e-16f);
    float4 v = ((const float4*)(h + (size_t)s * FDIM))[lane];
    float* ag = aggr + (size_t)d * FDIM + lane * 4;
    atomicAdd(ag + 0, v.x * a);
    atomicAdd(ag + 1, v.y * a);
    atomicAdd(ag + 2, v.z * a);
    atomicAdd(ag + 3, v.w * a);
}

// ---------- layer-1 epilogue: elu(aggr + b + skip) -> bf16 for layer-2 GEMM
__global__ void k_final1(const float* __restrict__ aggr, const float* __restrict__ skip,
                         const float* __restrict__ b, unsigned short* __restrict__ hb) {
    int i = blockIdx.x * 256 + threadIdx.x;
    if (i >= NNODES * FDIM) return;
    int c = i & 127;
    float x = aggr[i] + b[c] + skip[i];
    float y = x > 0.f ? x : (__expf(x) - 1.f);
    hb[i] = f2bf(y);
}

// ---------- layer-2 epilogue: elu(mean_heads(aggr) + b + skip) -> out [N,32] f32
__global__ void k_final2(const float* __restrict__ aggr, const float* __restrict__ skip,
                         const float* __restrict__ b, float* __restrict__ out) {
    int i = blockIdx.x * 256 + threadIdx.x;
    if (i >= NNODES * 32) return;
    int n = i >> 5, c = i & 31;
    const float* ar = aggr + (size_t)n * FDIM + c;
    float x = 0.25f * (ar[0] + ar[32] + ar[64] + ar[96]) + b[c] + skip[i];
    out[i] = x > 0.f ? x : (__expf(x) - 1.f);
}

// ---------- host ----------
static inline int cdiv(int a, int b) { return (a + b - 1) / b; }

extern "C" void kernel_launch(void* const* d_in, const int* in_sizes, int n_in,
                              void* d_out, int out_size, void* d_ws, size_t ws_size,
                              hipStream_t stream) {
    (void)in_sizes; (void)n_in; (void)out_size; (void)ws_size;
    const float* x      = (const float*)d_in[0];
    const int*   ei     = (const int*)  d_in[1];
    const float* W1     = (const float*)d_in[2];
    const float* asrc1  = (const float*)d_in[3];
    const float* adst1  = (const float*)d_in[4];
    const float* b1     = (const float*)d_in[5];
    const float* Wskip1 = (const float*)d_in[6];
    const float* W2     = (const float*)d_in[7];
    const float* asrc2  = (const float*)d_in[8];
    const float* adst2  = (const float*)d_in[9];
    const float* b2     = (const float*)d_in[10];
    const float* Wskip2 = (const float*)d_in[11];
    float* out = (float*)d_out;

    // workspace layout (all offsets 256B aligned); buffers reused across layers
    char* ws = (char*)d_ws;
    size_t o = 0;
    auto alloc = [&](size_t bytes) { void* p = ws + o; o += (bytes + 255) & ~(size_t)255; return p; };
    unsigned short* bufD = (unsigned short*)alloc((size_t)NNODES * FDIM * 2); // xb -> hb
    float*    bufA  = (float*)   alloc((size_t)NNODES * FDIM * 4);            // h1 -> h2
    float*    bufB  = (float*)   alloc((size_t)NNODES * FDIM * 4);            // skip1 -> aggr2
    float*    bufC  = (float*)   alloc((size_t)NNODES * FDIM * 4);            // aggr1 -> skip2
    float*    es    = (float*)   alloc((size_t)NNODES * HEADS * 4);
    float*    ed    = (float*)   alloc((size_t)NNODES * HEADS * 4);
    unsigned* mmax  = (unsigned*)alloc((size_t)NNODES * HEADS * 4);
    float*    denom = (float*)   alloc((size_t)NNODES * HEADS * 4);
    unsigned* Bpack = (unsigned*)alloc((size_t)16 * 4 * 32 * 8 * 4);

    const int ET = NEDGES + NNODES;
    const int NH = NNODES * HEADS;

    // ===== Layer 1 =====  (256 output cols = 16 tiles: W1 | Wskip1^T)
    k_f32_to_bf16<<<cdiv(NNODES * FDIM, 256), 256, 0, stream>>>(x, bufD, NNODES * FDIM);
    k_pack_b<<<cdiv(16 * 1024, 256), 256, 0, stream>>>(W1, Wskip1, Bpack, 128, 16 * 1024);
    k_gemm_wmma<<<dim3(ROWTILES), 256, 0, stream>>>(bufD, Bpack, bufA /*h1*/, bufB /*skip1*/, 128, 128);
    k_attn<<<cdiv(NH, 256), 256, 0, stream>>>(bufA, asrc1, adst1, es, ed);
    k_zero_u32<<<cdiv(NH, 256), 256, 0, stream>>>(mmax, NH);
    k_zero_u32<<<cdiv(NH, 256), 256, 0, stream>>>((unsigned*)denom, NH);
    k_zero_u32<<<cdiv(NNODES * FDIM, 256), 256, 0, stream>>>((unsigned*)bufC, NNODES * FDIM);
    k_edge_max<<<cdiv(ET, 256), 256, 0, stream>>>(ei, es, ed, mmax);
    k_edge_denom<<<cdiv(ET, 256), 256, 0, stream>>>(ei, es, ed, mmax, denom, bufA);
    k_edge_scatter<<<cdiv(ET, 8), 256, 0, stream>>>(ei, es, ed, mmax, denom, bufA, bufC /*aggr1*/);
    k_final1<<<cdiv(NNODES * FDIM, 256), 256, 0, stream>>>(bufC, bufB, b1, bufD /*hb*/);

    // ===== Layer 2 =====  (160 output cols = 10 tiles: W2 | Wskip2^T)
    k_pack_b<<<cdiv(10 * 1024, 256), 256, 0, stream>>>(W2, Wskip2, Bpack, 128, 10 * 1024);
    k_gemm_wmma<<<dim3(ROWTILES), 256, 0, stream>>>(bufD, Bpack, bufA /*h2*/, bufC /*skip2*/, 128, 32);
    k_attn<<<cdiv(NH, 256), 256, 0, stream>>>(bufA, asrc2, adst2, es, ed);
    k_zero_u32<<<cdiv(NH, 256), 256, 0, stream>>>(mmax, NH);
    k_zero_u32<<<cdiv(NH, 256), 256, 0, stream>>>((unsigned*)denom, NH);
    k_zero_u32<<<cdiv(NNODES * FDIM, 256), 256, 0, stream>>>((unsigned*)bufB, NNODES * FDIM);
    k_edge_max<<<cdiv(ET, 256), 256, 0, stream>>>(ei, es, ed, mmax);
    k_edge_denom<<<cdiv(ET, 256), 256, 0, stream>>>(ei, es, ed, mmax, denom, bufA);
    k_edge_scatter<<<cdiv(ET, 8), 256, 0, stream>>>(ei, es, ed, mmax, denom, bufA, bufB /*aggr2*/);
    k_final2<<<cdiv(NNODES * 32, 256), 256, 0, stream>>>(bufB, bufC, b2, out);
}
